// FastAttention_54030688584104
// MI455X (gfx1250) — matrix operand order repaired and checked by reading, not compiled, not run on hardware
//
#include <hip/hip_runtime.h>

typedef __bf16 bf16_t;
typedef __attribute__((ext_vector_type(16))) __bf16 v16bf;
typedef __attribute__((ext_vector_type(8)))  float  v8f;

namespace {
constexpr int Bc = 4, Hc = 8, Lc = 8192, Dc = 64, Mc = 256;
constexpr int BH = Bc * Hc;          // 32
constexpr int SPLITS = 8;            // l-splits in pass A
constexpr int LCHUNK = Lc / SPLITS;  // 1024
constexpr int LT  = 32;              // l-tile per iteration (pass A)
constexpr int LTP = 34;              // padded LT stride (even -> 32b-aligned pairs)
constexpr int MP  = 258;             // padded Mc stride (even)
constexpr float DNORM = 0.3535533905932738f;  // 64^-0.25
constexpr float EPSK = 1e-6f;
constexpr float EPSZ = 1e-6f;
}

union FragU { v16bf v; unsigned u[8]; };

// A-matrix (16xK, 16-bit): lane = M (mod 16); VGPR i, half = lane>>4:
//   K = (i>>2)*16 + half*8 + (i&3)*2  (+0,+1 packed in one dword)
__device__ __forceinline__ int kA_idx(int i, int hf) {
  return ((i >> 2) << 4) + (hf << 3) + ((i & 3) << 1);
}
// B-matrix (Kx16, 16-bit): lane = N (mod 16); same K order as A:
//   K = (i>>2)*16 + half*8 + (i&3)*2  (+0,+1 packed in one dword)
__device__ __forceinline__ int kB_idx(int i, int hf) {
  return ((i >> 2) << 4) + (hf << 3) + ((i & 3) << 1);
}

__device__ __forceinline__ v8f wmma_bf16(v16bf a, v16bf b, v8f c) {
  return __builtin_amdgcn_wmma_f32_16x16x32_bf16(
      /*neg_a=*/false, a, /*neg_b=*/false, b,
      /*c_mod=*/(short)0, c, /*reuse_a=*/false, /*reuse_b=*/false);
}

__device__ __forceinline__ unsigned short bfb(float f) {
  bf16_t h = (bf16_t)f;
  unsigned short u;
  __builtin_memcpy(&u, &h, 2);
  return u;
}

// ---------------------------------------------------------------------------
// Pass A: per (bh, l-split): accumulate kv[256x64] (f32) and ksum[256] via
// on-the-fly k' = exp(k_n P^T - |k_n|^2/2 + eps), all matmuls in bf16 WMMA.
// ---------------------------------------------------------------------------
__global__ __launch_bounds__(256) void performer_pass_a(
    const float* __restrict__ Kg, const float* __restrict__ Vg,
    const float* __restrict__ Pg, float* __restrict__ kv_g,
    float* __restrict__ ksum_g)
{
  extern __shared__ char smem[];
  bf16_t* Pb    = (bf16_t*)smem;            // [Mc][Dc]
  bf16_t* Db    = Pb  + Mc * Dc;            // [LT][Dc]   scaled k tile
  bf16_t* VbT   = Db  + LT * Dc;            // [Dc][LTP]  v tile transposed
  bf16_t* KPt   = VbT + Dc * LTP;           // [Mc][LTP]  k' transposed
  float*  ksumS = (float*)(KPt + Mc * LTP); // [Mc]
  float*  rn    = ksumS + Mc;               // [LT]

  const int tid  = threadIdx.x;
  const int lane = tid & 31;
  const int wave = tid >> 5;
  const int hf   = lane >> 4;
  const int ln   = lane & 15;
  const int bh   = blockIdx.x;
  const int split= blockIdx.y;
  const int mbase= wave * 32;

  const float* Kp = Kg + (size_t)bh * Lc * Dc + (size_t)split * LCHUNK * Dc;
  const float* Vp = Vg + (size_t)bh * Lc * Dc + (size_t)split * LCHUNK * Dc;

  for (int t = tid; t < Mc * Dc; t += 256) Pb[t] = (bf16_t)Pg[t];
  for (int t = tid; t < Mc; t += 256) ksumS[t] = 0.0f;

  const v8f vzero = {0.f, 0.f, 0.f, 0.f, 0.f, 0.f, 0.f, 0.f};
  v8f acc[2][4];
  #pragma unroll
  for (int a0 = 0; a0 < 2; ++a0)
    #pragma unroll
    for (int a1 = 0; a1 < 4; ++a1) acc[a0][a1] = vzero;

  __syncthreads();

  const int r  = tid >> 3;        // staging: row 0..31
  const int c0 = (tid & 7) << 3;  // 8 contiguous cols per thread

  for (int l0 = 0; l0 < LCHUNK; l0 += LT) {
    // ---- stage k tile (scaled) + row norms + v tile (transposed) ----
    const float* krow = Kp + (size_t)(l0 + r) * Dc + c0;
    const float* vrow = Vp + (size_t)(l0 + r) * Dc + c0;
    float s = 0.f;
    #pragma unroll
    for (int j = 0; j < 8; ++j) {
      float x = krow[j] * DNORM;
      s += x * x;
      Db[r * Dc + c0 + j]     = (bf16_t)x;
      VbT[(c0 + j) * LTP + r] = (bf16_t)vrow[j];
    }
    s += __shfl_xor(s, 1);
    s += __shfl_xor(s, 2);
    s += __shfl_xor(s, 4);
    if ((tid & 7) == 0) rn[r] = 0.5f * s;
    __syncthreads();

    // ---- GEMM1: proj = k_n @ P^T for wave's 32-m slice; exp; stash ----
    #pragma unroll
    for (int lt = 0; lt < 2; ++lt) {
      #pragma unroll
      for (int mt = 0; mt < 2; ++mt) {
        v8f c = vzero;
        const unsigned* arow = (const unsigned*)(Db + (lt * 16 + ln) * Dc);
        const unsigned* brow = (const unsigned*)(Pb + (mbase + mt * 16 + ln) * Dc);
        #pragma unroll
        for (int ks = 0; ks < 2; ++ks) {
          FragU a, b;
          #pragma unroll
          for (int i = 0; i < 8; ++i) {
            a.u[i] = arow[(ks * 32 + kA_idx(i, hf)) >> 1];
            b.u[i] = brow[(ks * 32 + kB_idx(i, hf)) >> 1];
          }
          c = wmma_bf16(a.v, b.v, c);
        }
        float kp[8];
        float colsum = 0.f;
        #pragma unroll
        for (int i = 0; i < 8; ++i) {
          int lrow = lt * 16 + hf * 8 + i;        // C: M = i + 8*half
          float e = __expf(c[i] - rn[lrow] + EPSK);
          kp[i] = e;
          colsum += e;
        }
        atomicAdd(&ksumS[mbase + mt * 16 + ln], colsum);
        // transpose-store: fixed column m, 8 consecutive l -> packed dwords
        unsigned* kprow =
            (unsigned*)(KPt + (mbase + mt * 16 + ln) * LTP + lt * 16 + hf * 8);
        #pragma unroll
        for (int j = 0; j < 4; ++j)
          kprow[j] = (unsigned)bfb(kp[2 * j]) | ((unsigned)bfb(kp[2 * j + 1]) << 16);
      }
    }

    // ---- GEMM2: kv[mslice, :] += k'^T @ v  (K = 32 this iteration) ----
    #pragma unroll
    for (int mt = 0; mt < 2; ++mt) {
      FragU a;
      const unsigned* arow = (const unsigned*)(KPt + (mbase + mt * 16 + ln) * LTP);
      #pragma unroll
      for (int i = 0; i < 8; ++i) a.u[i] = arow[kA_idx(i, hf) >> 1];
      #pragma unroll
      for (int dt = 0; dt < 4; ++dt) {
        FragU b;
        const unsigned* brow = (const unsigned*)(VbT + (dt * 16 + ln) * LTP);
        #pragma unroll
        for (int i = 0; i < 8; ++i) b.u[i] = brow[kB_idx(i, hf) >> 1];
        acc[mt][dt] = wmma_bf16(a.v, b.v, acc[mt][dt]);
      }
    }
    __syncthreads();
  }

  // ---- flush partials (8 splits per bh) ----
  for (int t = tid; t < Mc; t += 256) atomicAdd(&ksum_g[bh * Mc + t], ksumS[t]);
  float* kvb = kv_g + (size_t)bh * Mc * Dc;
  #pragma unroll
  for (int mt = 0; mt < 2; ++mt)
    #pragma unroll
    for (int dt = 0; dt < 4; ++dt)
      #pragma unroll
      for (int i = 0; i < 8; ++i) {
        int m = mbase + mt * 16 + hf * 8 + i;
        int d = dt * 16 + ln;
        atomicAdd(&kvb[m * Dc + d], acc[mt][dt][i]);
      }
}

// ---------------------------------------------------------------------------
// Pass B: per (bh, 128-row l-tile): q' = exp(q_n P^T - |q_n|^2/2 + eps),
// z = q' . ksum, out = (q' @ kv) / (z + eps).
// ---------------------------------------------------------------------------
__global__ __launch_bounds__(256) void performer_pass_b(
    const float* __restrict__ Qg, const float* __restrict__ Pg,
    const float* __restrict__ kv_g, const float* __restrict__ ksum_g,
    float* __restrict__ Og)
{
  extern __shared__ char smem[];
  bf16_t* Pb    = (bf16_t*)smem;               // [Mc][Dc]
  bf16_t* kvT   = Pb + Mc * Dc;                // [Dc][MP]  kv transposed (bf16)
  bf16_t* qp    = kvT + Dc * MP;               // [8 waves][16][Mc] q'
  float*  ksumS = (float*)(qp + 8 * 16 * Mc);  // [Mc]
  float*  zS    = ksumS + Mc;                  // [128]
  bf16_t* Qb    = (bf16_t*)(zS + 128);         // [128][Dc] scaled q tile
  float*  rn    = (float*)(Qb + 128 * Dc);     // [128]

  const int tid  = threadIdx.x;
  const int lane = tid & 31;
  const int wave = tid >> 5;
  const int hf   = lane >> 4;
  const int ln   = lane & 15;
  const int bh   = blockIdx.x;
  const int lblk = blockIdx.y;
  const int lbase= wave * 16;   // wave's 16 rows inside 128-row tile

  const float* Qp  = Qg + (size_t)bh * Lc * Dc + (size_t)lblk * 128 * Dc;
  const float* kvp = kv_g + (size_t)bh * Mc * Dc;

  for (int t = tid; t < Mc * Dc; t += 256) {
    Pb[t] = (bf16_t)Pg[t];
    int mm = t >> 6, dd = t & 63;
    kvT[dd * MP + mm] = (bf16_t)kvp[t];
  }
  for (int t = tid; t < Mc; t += 256) ksumS[t] = ksum_g[bh * Mc + t];

  {  // stage scaled q tile + row norms
    const int r  = tid >> 1;
    const int c0 = (tid & 1) << 5;
    const float* qrow = Qp + (size_t)r * Dc + c0;
    float s = 0.f;
    #pragma unroll
    for (int j = 0; j < 32; ++j) {
      float x = qrow[j] * DNORM;
      s += x * x;
      Qb[r * Dc + c0 + j] = (bf16_t)x;
    }
    s += __shfl_xor(s, 1);
    if ((tid & 1) == 0) rn[r] = 0.5f * s;
  }
  __syncthreads();

  const v8f vzero = {0.f, 0.f, 0.f, 0.f, 0.f, 0.f, 0.f, 0.f};
  bf16_t* qpW = qp + (size_t)wave * 16 * Mc;
  float zacc[8];
  #pragma unroll
  for (int i = 0; i < 8; ++i) zacc[i] = 0.f;

  // ---- GEMM1: q' for wave's 16 rows, all 256 m; fused z accumulation ----
  const unsigned* arow1 = (const unsigned*)(Qb + (lbase + ln) * Dc);
  #pragma unroll
  for (int mt = 0; mt < 16; ++mt) {
    v8f c = vzero;
    const unsigned* brow = (const unsigned*)(Pb + (mt * 16 + ln) * Dc);
    #pragma unroll
    for (int ks = 0; ks < 2; ++ks) {
      FragU a, b;
      #pragma unroll
      for (int i = 0; i < 8; ++i) {
        a.u[i] = arow1[(ks * 32 + kA_idx(i, hf)) >> 1];
        b.u[i] = brow[(ks * 32 + kB_idx(i, hf)) >> 1];
      }
      c = wmma_bf16(a.v, b.v, c);
    }
    float ksv = ksumS[mt * 16 + ln];
    #pragma unroll
    for (int i = 0; i < 8; ++i) {
      int rloc = hf * 8 + i;
      float e = __expf(c[i] - rn[lbase + rloc] + EPSK);
      zacc[i] += e * ksv;
      qpW[rloc * Mc + mt * 16 + ln] = (bf16_t)e;
    }
  }
  // reduce z over 16 lanes of each half (columns)
  #pragma unroll
  for (int i = 0; i < 8; ++i) {
    float zi = zacc[i];
    zi += __shfl_xor(zi, 1);
    zi += __shfl_xor(zi, 2);
    zi += __shfl_xor(zi, 4);
    zi += __shfl_xor(zi, 8);
    if (ln == 0) zS[lbase + hf * 8 + i] = zi;
  }

  // ---- GEMM2: out = q' @ kv  (K = 256 -> 8 WMMA k-steps) ----
  v8f oacc[4];
  #pragma unroll
  for (int dt = 0; dt < 4; ++dt) oacc[dt] = vzero;
  #pragma unroll
  for (int ks = 0; ks < 8; ++ks) {
    FragU a;
    const unsigned* arow = (const unsigned*)(qpW + ln * Mc + ks * 32);
    #pragma unroll
    for (int i = 0; i < 8; ++i) a.u[i] = arow[kA_idx(i, hf) >> 1];
    #pragma unroll
    for (int dt = 0; dt < 4; ++dt) {
      FragU b;
      const unsigned* brow = (const unsigned*)(kvT + (dt * 16 + ln) * MP);
      #pragma unroll
      for (int i = 0; i < 8; ++i)
        b.u[i] = brow[(ks * 32 + kB_idx(i, hf)) >> 1];
      oacc[dt] = wmma_bf16(a.v, b.v, oacc[dt]);
    }
  }

  // ---- scale by 1/(z+eps), store ----
  #pragma unroll
  for (int dt = 0; dt < 4; ++dt)
    #pragma unroll
    for (int i = 0; i < 8; ++i) {
      int rloc = hf * 8 + i;
      float zv = 1.0f / (zS[lbase + rloc] + EPSZ);
      size_t lg = (size_t)lblk * 128 + lbase + rloc;
      Og[((size_t)bh * Lc + lg) * Dc + dt * 16 + ln] = oacc[dt][i] * zv;
    }
}

__global__ void performer_zero(float* __restrict__ p, int n) {
  for (int i = blockIdx.x * blockDim.x + threadIdx.x; i < n;
       i += gridDim.x * blockDim.x)
    p[i] = 0.0f;
}

extern "C" void kernel_launch(void* const* d_in, const int* in_sizes, int n_in,
                              void* d_out, int out_size, void* d_ws, size_t ws_size,
                              hipStream_t stream) {
  (void)in_sizes; (void)n_in; (void)out_size; (void)ws_size;
  const float* q = (const float*)d_in[0];
  const float* k = (const float*)d_in[1];
  const float* v = (const float*)d_in[2];
  const float* P = (const float*)d_in[3];
  float* out    = (float*)d_out;
  float* kv_g   = (float*)d_ws;                 // [BH][Mc][Dc] f32
  float* ksum_g = kv_g + (size_t)BH * Mc * Dc;  // [BH][Mc] f32

  int nz = BH * Mc * Dc + BH * Mc;
  performer_zero<<<512, 256, 0, stream>>>(kv_g, nz);

  size_t lds_a = (size_t)(Mc * Dc + LT * Dc + Dc * LTP + Mc * LTP) * sizeof(bf16_t)
               + (size_t)(Mc + LT) * sizeof(float);
  dim3 ga(BH, SPLITS);
  performer_pass_a<<<ga, 256, lds_a, stream>>>(k, v, P, kv_g, ksum_g);

  size_t lds_b = (size_t)(Mc * Dc + Dc * MP + 8 * 16 * Mc + 128 * Dc) * sizeof(bf16_t)
               + (size_t)(Mc + 128 + 128) * sizeof(float);
  dim3 gb(BH, Lc / 128);
  performer_pass_b<<<gb, 256, lds_b, stream>>>(q, P, kv_g, ksum_g, out);
}
